// DolmaGPT_86182813762108
// MI455X (gfx1250) — matrix operand order repaired and checked by reading, not compiled, not run on hardware
//
#include <hip/hip_runtime.h>

// ---------------------------------------------------------------------------
// DolmaGPT forward for MI455X (gfx1250, wave32, WMMA).
// Memory-bound problem (~340MB weights, ~375 GFLOP): bf16 weights/activations
// through v_wmma_f32_16x16x32_bf16, fp32 residual stream, async global->LDS
// (ASYNCcnt) double-buffered tile pipeline.
// ---------------------------------------------------------------------------

typedef __attribute__((ext_vector_type(16))) __bf16 bf16x16;
typedef __attribute__((ext_vector_type(8)))  float  f32x8;
typedef __attribute__((address_space(3)))    __bf16 lds_bf16_t;

union FragBF { uint4 q[2]; bf16x16 v; };

// model constants
constexpr int kD   = 1024;
constexpr int kH   = 16;
constexpr int kHD  = 64;
constexpr int kT   = 1024;
constexpr int kB   = 2;
constexpr int kL   = 4;
constexpr int kV   = 32000;
constexpr int kMLP = 4096;
constexpr int kM   = kB * kT;   // 2048 token rows

__device__ __forceinline__ float gelu_f(float x) {
  return 0.5f * x * (1.0f + erff(x * 0.70710678118654752f));
}

// CDNA5 async global->LDS copy, 16 bytes per lane, tracked by ASYNCcnt.
__device__ __forceinline__ void async_copy16(lds_bf16_t* dst, const __bf16* src) {
  asm volatile("global_load_async_to_lds_b128 %0, %1, off"
               :: "v"(dst), "v"(src)
               : "memory");
}
__device__ __forceinline__ void wait_async0() {
  asm volatile("s_wait_asynccnt 0x0" ::: "memory");
}

// ---------------------------------------------------------------------------
// f32 -> bf16 bulk convert (weights)
// ---------------------------------------------------------------------------
__global__ void cvt_f32_to_bf16_kernel(const float* __restrict__ in,
                                       __bf16* __restrict__ out, long n) {
  long i = (long)blockIdx.x * blockDim.x + threadIdx.x;
  const long stride = (long)gridDim.x * blockDim.x;
  for (; i < n; i += stride) out[i] = (__bf16)in[i];
}

// ---------------------------------------------------------------------------
// x = wte[input_ids]   (fp32 residual stream)
// ---------------------------------------------------------------------------
__global__ void embed_kernel(const int* __restrict__ ids,
                             const float* __restrict__ wte,
                             float* __restrict__ x) {
  const int row = blockIdx.x;
  const long id = (long)ids[row];
  const float* src = wte + id * kD;
  float* dst = x + (long)row * kD;
  for (int c = threadIdx.x; c < kD; c += blockDim.x) dst[c] = src[c];
}

// ---------------------------------------------------------------------------
// LayerNorm (fp32 in, bf16 out), one block per row, 256 threads
// ---------------------------------------------------------------------------
__global__ void layernorm_kernel(const float* __restrict__ x,
                                 const float* __restrict__ g,
                                 const float* __restrict__ b,
                                 __bf16* __restrict__ out, int n) {
  __shared__ float red[256];
  const int row = blockIdx.x;
  const int tid = threadIdx.x;
  const float* xr = x + (long)row * n;

  float s = 0.f;
  for (int c = tid; c < n; c += blockDim.x) s += xr[c];
  red[tid] = s; __syncthreads();
  for (int w = 128; w > 0; w >>= 1) { if (tid < w) red[tid] += red[tid + w]; __syncthreads(); }
  const float mean = red[0] / (float)n;
  __syncthreads();

  float v = 0.f;
  for (int c = tid; c < n; c += blockDim.x) { float d = xr[c] - mean; v += d * d; }
  red[tid] = v; __syncthreads();
  for (int w = 128; w > 0; w >>= 1) { if (tid < w) red[tid] += red[tid + w]; __syncthreads(); }
  const float rstd = rsqrtf(red[0] / (float)n + 1e-5f);

  __bf16* o = out + (long)row * n;
  for (int c = tid; c < n; c += blockDim.x)
    o[c] = (__bf16)((xr[c] - mean) * rstd * g[c] + b[c]);
}

// ---------------------------------------------------------------------------
// Split bf16 qkv [M, 3D] into q,k [B,H,T,HD] and v transposed [B,H,HD,T]
// ---------------------------------------------------------------------------
__global__ void split_qkv_kernel(const __bf16* __restrict__ qkv,
                                 __bf16* __restrict__ q,
                                 __bf16* __restrict__ k,
                                 __bf16* __restrict__ vt) {
  const long idx = (long)blockIdx.x * blockDim.x + threadIdx.x; // [b,h,t,d]
  const int d = (int)(idx & 63);
  const int t = (int)((idx >> 6) & 1023);
  const int h = (int)((idx >> 16) & 15);
  const int b = (int)(idx >> 20);
  const long src = ((long)(b * kT + t)) * (3 * kD) + h * kHD + d;
  q[idx] = qkv[src];
  k[idx] = qkv[src + kD];
  vt[(((long)(b * kH + h)) * kHD + d) * kT + t] = qkv[src + 2 * kD];
}

// ---------------------------------------------------------------------------
// Row-wise causal softmax, in place on bf16 scores [Z, T, T].
// Writes zeros for j > i so the AV GEMM needs no masking.
// ---------------------------------------------------------------------------
__global__ void softmax_causal_kernel(__bf16* __restrict__ s) {
  __shared__ float red[256];
  const int i = blockIdx.x & (kT - 1);
  const long z = blockIdx.x >> 10;
  __bf16* row = s + (z * kT + i) * (long)kT;
  const int tid = threadIdx.x;
  const int nvalid = i + 1;

  float vals[kT / 256];
  float mx = -3.0e38f;
#pragma unroll
  for (int c = 0; c < kT / 256; ++c) {
    int j = tid + c * 256;
    vals[c] = (j < nvalid) ? (float)row[j] : -3.0e38f;
    mx = fmaxf(mx, vals[c]);
  }
  red[tid] = mx; __syncthreads();
  for (int w = 128; w > 0; w >>= 1) { if (tid < w) red[tid] = fmaxf(red[tid], red[tid + w]); __syncthreads(); }
  const float rowmax = red[0];
  __syncthreads();

  float sum = 0.f;
#pragma unroll
  for (int c = 0; c < kT / 256; ++c) {
    int j = tid + c * 256;
    float e = (j < nvalid) ? __expf(vals[c] - rowmax) : 0.f;
    vals[c] = e; sum += e;
  }
  red[tid] = sum; __syncthreads();
  for (int w = 128; w > 0; w >>= 1) { if (tid < w) red[tid] += red[tid + w]; __syncthreads(); }
  const float inv = 1.0f / red[0];
#pragma unroll
  for (int c = 0; c < kT / 256; ++c) {
    int j = tid + c * 256;
    row[j] = (__bf16)(vals[c] * inv);
  }
}

// ---------------------------------------------------------------------------
// Tiled bf16 GEMM:  C[z] = A[z] (MxK, row-major) * B[z]^T (B is NxK row-major)
// Wave tile 32x32 (2x2 wmma f32 accum), block tile (WAVES_M*32)x(WAVES_N*32).
// Double-buffered LDS tiles filled with global_load_async_to_lds_b128
// (ASYNCcnt); rows padded to 40 bf16 (80B) against bank conflicts.
// Fragment layouts per CDNA5 ISA 7.12.2:
//   A frag: lane(0-15)=m, lane>>4 selects K half; pairs {h*8..h*8+7, 16+h*8..}
//   B frag: lane(0-15)=n, lane>>4 selects K half; pairs {h*8..h*8+7, 16+h*8..}
//   C/D   : lane(0-15)=n, vgpr r -> m = (lane>>4)*8 + r
// flags: bit0 = GELU, bit1 = attention-scores epilogue (scale + ALiBi + causal)
// ---------------------------------------------------------------------------
template <int WAVES_M, int WAVES_N>
__global__ __launch_bounds__(WAVES_M * WAVES_N * 32)
void gemm_bf16_kernel(const __bf16* __restrict__ A, int lda, long aStrideZ,
                      const __bf16* __restrict__ Bm, int ldb, long bStrideZ,
                      float* Cf, __bf16* Cb, int ldc,
                      long cStrideOuter, long cStrideInner, int batchInner,
                      const float* __restrict__ bias,
                      const float* residual,  // aliases Cf when used: no restrict
                      int M, int N, int K, int flags, float scaleP) {
  constexpr int BM  = WAVES_M * 32;
  constexpr int BN  = WAVES_N * 32;
  constexpr int NT  = WAVES_M * WAVES_N * 32;
  constexpr int KT  = 32;
  constexpr int LDP = KT + 8;            // padded LDS row, bf16 elems
  constexpr int CA  = (BM * 4) / NT;     // 16B chunks per thread (A tile)
  constexpr int CB  = (BN * 4) / NT;

  const int m0 = blockIdx.y * BM;
  const int n0 = blockIdx.x * BN;
  if ((flags & 2) && (n0 > m0 + BM - 1)) return;  // block fully above diagonal

  __shared__ __bf16 lds_a[2][BM * LDP];
  __shared__ __bf16 lds_b[2][BN * LDP];

  const int z = blockIdx.z;
  const __bf16* Ag = A  + (long)z * aStrideZ + (long)m0 * lda;
  const __bf16* Bg = Bm + (long)z * bStrideZ + (long)n0 * ldb;

  const int tid  = threadIdx.x;
  const int lane = tid & 31;
  const int wave = tid >> 5;
  const int wm   = (wave / WAVES_N) * 32;
  const int wn   = (wave % WAVES_N) * 32;
  const int l15  = lane & 15;
  const int lh   = lane >> 4;

  f32x8 acc[2][2];
#pragma unroll
  for (int i = 0; i < 2; ++i)
#pragma unroll
    for (int j = 0; j < 2; ++j)
#pragma unroll
      for (int r = 0; r < 8; ++r) acc[i][j][r] = 0.f;

  // issue one K-tile's async copies (16B per lane per chunk)
  auto issueTile = [&](int k0, int buf) {
#pragma unroll
    for (int i = 0; i < CA; ++i) {
      int t = tid + i * NT; int r = t >> 2; int c = (t & 3) * 8;
      async_copy16((lds_bf16_t*)&lds_a[buf][r * LDP + c],
                   Ag + (long)r * lda + k0 + c);
    }
#pragma unroll
    for (int i = 0; i < CB; ++i) {
      int t = tid + i * NT; int r = t >> 2; int c = (t & 3) * 8;
      async_copy16((lds_bf16_t*)&lds_b[buf][r * LDP + c],
                   Bg + (long)r * ldb + k0 + c);
    }
  };

  issueTile(0, 0);
  wait_async0();
  __syncthreads();

  const int nK = K / KT;
  for (int kt = 0; kt < nK; ++kt) {
    const int cur = kt & 1;
    // prefetch next tile into the other buffer; it overlaps the WMMAs below.
    // (that buffer was last read at step kt-1, protected by the barrier there)
    if (kt + 1 < nK) issueTile((kt + 1) * KT, cur ^ 1);

    FragBF af[2], bf[2];
#pragma unroll
    for (int mt = 0; mt < 2; ++mt) {
      const __bf16* p = &lds_a[cur][(wm + mt * 16 + l15) * LDP + lh * 8];
      af[mt].q[0] = *(const uint4*)p;           // K = h*8 .. h*8+7
      af[mt].q[1] = *(const uint4*)(p + 16);    // K = 16+h*8 .. 16+h*8+7
    }
#pragma unroll
    for (int nt = 0; nt < 2; ++nt) {
      const __bf16* p = &lds_b[cur][(wn + nt * 16 + l15) * LDP + lh * 8];
      bf[nt].q[0] = *(const uint4*)p;           // K = h*8 .. h*8+7
      bf[nt].q[1] = *(const uint4*)(p + 16);    // K = 16+h*8 .. 16+h*8+7
    }
#pragma unroll
    for (int mt = 0; mt < 2; ++mt)
#pragma unroll
      for (int nt = 0; nt < 2; ++nt)
        acc[mt][nt] = __builtin_amdgcn_wmma_f32_16x16x32_bf16(
            false, af[mt].v, false, bf[nt].v, (short)0, acc[mt][nt],
            false, false);

    // only the (kt+1) tile's copies are outstanding: wait out the residual
    // latency not hidden by the compute above, then make it block-visible.
    wait_async0();
    __syncthreads();
  }

  // ---- epilogue ----
  const long zc = (long)(z / batchInner) * cStrideOuter +
                  (long)(z % batchInner) * cStrideInner;
  const bool scoresMode = (flags & 2) != 0;
  float slope = 0.f;
  if (scoresMode) {
    const int hIdx = z % batchInner;                 // head index
    slope = exp2f(-0.5f * (float)(hIdx + 1));        // 2^-(m), m=(h+1)*8/H
  }
#pragma unroll
  for (int mt = 0; mt < 2; ++mt) {
#pragma unroll
    for (int nt = 0; nt < 2; ++nt) {
      const int n = n0 + wn + nt * 16 + l15;
#pragma unroll
      for (int r = 0; r < 8; ++r) {
        const int m = m0 + wm + mt * 16 + lh * 8 + r;
        float v = acc[mt][nt][r];
        if (scoresMode) {
          if (n <= m)
            Cb[zc + (long)m * ldc + n] =
                (__bf16)(v * scaleP - slope * (float)(m - n));
          continue;
        }
        if (bias) v += bias[n];
        if (flags & 1) v = gelu_f(v);
        const long idx = zc + (long)m * ldc + n;
        if (residual) v += residual[idx];
        if (Cb) Cb[idx] = (__bf16)v;
        else    Cf[idx] = v;
      }
    }
  }
}

// ---------------------------------------------------------------------------
// Host-side orchestration
// ---------------------------------------------------------------------------
extern "C" void kernel_launch(void* const* d_in, const int* in_sizes, int n_in,
                              void* d_out, int out_size, void* d_ws, size_t ws_size,
                              hipStream_t stream) {
  (void)in_sizes; (void)n_in; (void)out_size; (void)ws_size;
  const int*   ids   = (const int*)  d_in[0];
  const float* wte   = (const float*)d_in[1];
  const float* ln1g  = (const float*)d_in[2];
  const float* ln1b  = (const float*)d_in[3];
  const float* attnw = (const float*)d_in[4];
  const float* attnb = (const float*)d_in[5];
  const float* projw = (const float*)d_in[6];
  const float* projb = (const float*)d_in[7];
  const float* ln2g  = (const float*)d_in[8];
  const float* ln2b  = (const float*)d_in[9];
  const float* fcw   = (const float*)d_in[10];
  const float* fcb   = (const float*)d_in[11];
  const float* fc2w  = (const float*)d_in[12];
  const float* fc2b  = (const float*)d_in[13];
  const float* lnfg  = (const float*)d_in[14];
  const float* lnfb  = (const float*)d_in[15];
  const float* lmw   = (const float*)d_in[16];
  float* logits = (float*)d_out;

  char* ws = (char*)d_ws;
  size_t off = 0;
  auto take = [&](size_t bytes) -> void* {
    void* p = ws + off;
    off += (bytes + 255) & ~(size_t)255;
    return p;
  };
  __bf16* w_attn = (__bf16*)take((size_t)kL * 3 * kD * kD * 2);
  __bf16* w_proj = (__bf16*)take((size_t)kL * kD * kD * 2);
  __bf16* w_fc   = (__bf16*)take((size_t)kL * kMLP * kD * 2);
  __bf16* w_fc2  = (__bf16*)take((size_t)kL * kD * kMLP * 2);
  __bf16* w_lm   = (__bf16*)take((size_t)kV * kD * 2);
  float*  x      = (float*) take((size_t)kM * kD * 4);
  __bf16* hb     = (__bf16*)take((size_t)kM * kD * 2);
  __bf16* qkvb   = (__bf16*)take((size_t)kM * 3 * kD * 2);
  __bf16* qb     = (__bf16*)take((size_t)kM * kD * 2);
  __bf16* kb     = (__bf16*)take((size_t)kM * kD * 2);
  __bf16* vtb    = (__bf16*)take((size_t)kM * kD * 2);
  __bf16* probs  = (__bf16*)take((size_t)kB * kH * kT * kT * 2);
  __bf16* y2b    = (__bf16*)take((size_t)kM * kD * 2);
  __bf16* mlpb   = (__bf16*)take((size_t)kM * kMLP * 2);

  // ---- weights -> bf16 (every call; no cross-call state) ----
  cvt_f32_to_bf16_kernel<<<4096, 256, 0, stream>>>(attnw, w_attn, (long)kL * 3 * kD * kD);
  cvt_f32_to_bf16_kernel<<<4096, 256, 0, stream>>>(projw, w_proj, (long)kL * kD * kD);
  cvt_f32_to_bf16_kernel<<<4096, 256, 0, stream>>>(fcw,  w_fc,  (long)kL * kMLP * kD);
  cvt_f32_to_bf16_kernel<<<4096, 256, 0, stream>>>(fc2w, w_fc2, (long)kL * kD * kMLP);
  cvt_f32_to_bf16_kernel<<<4096, 256, 0, stream>>>(lmw,  w_lm,  (long)kV * kD);

  // ---- embedding ----
  embed_kernel<<<kM, 256, 0, stream>>>(ids, wte, x);

  const float scale = 0.125f;  // 1/sqrt(HD=64)
  for (int l = 0; l < kL; ++l) {
    // LN1 -> bf16
    layernorm_kernel<<<kM, 256, 0, stream>>>(x, ln1g + l * kD, ln1b + l * kD, hb, kD);
    // qkv = h @ attn_w^T + attn_b  -> bf16 [M, 3D]
    gemm_bf16_kernel<2, 4><<<dim3(3 * kD / 128, kM / 64, 1), 256, 0, stream>>>(
        hb, kD, 0, w_attn + (size_t)l * 3 * kD * kD, kD, 0,
        nullptr, qkvb, 3 * kD, 0, 0, 1,
        attnb + l * 3 * kD, nullptr, kM, 3 * kD, kD, 0, 0.f);
    // split into q,k [B,H,T,HD] and v^T [B,H,HD,T]
    split_qkv_kernel<<<(kB * kH * kT * kHD) / 256, 256, 0, stream>>>(qkvb, qb, kb, vtb);
    // scores = q k^T * scale + alibi  (bf16, causal blocks only), batched over B*H
    gemm_bf16_kernel<2, 4><<<dim3(kT / 128, kT / 64, kB * kH), 256, 0, stream>>>(
        qb, kHD, (long)kT * kHD, kb, kHD, (long)kT * kHD,
        nullptr, probs, kT, (long)kH * kT * kT, (long)kT * kT, kH,
        nullptr, nullptr, kT, kT, kHD, /*flags=scores*/2, scale);
    // softmax (in place, zero-fills above diagonal)
    softmax_causal_kernel<<<kB * kH * kT, 256, 0, stream>>>(probs);
    // y = P @ V : A=probs [T,T], B=v^T rows d over T -> out [B,T,D] interleaved heads
    gemm_bf16_kernel<2, 2><<<dim3(kHD / 64, kT / 64, kB * kH), 128, 0, stream>>>(
        probs, kT, (long)kT * kT, vtb, kT, (long)kHD * kT,
        nullptr, y2b, kD, (long)kT * kD, kHD, kH,
        nullptr, nullptr, kT, kHD, kT, 0, 0.f);
    // x += y @ proj_w^T + proj_b   (fp32 residual)
    gemm_bf16_kernel<2, 4><<<dim3(kD / 128, kM / 64, 1), 256, 0, stream>>>(
        y2b, kD, 0, w_proj + (size_t)l * kD * kD, kD, 0,
        x, nullptr, kD, 0, 0, 1,
        projb + l * kD, x, kM, kD, kD, 0, 0.f);
    // LN2 -> bf16
    layernorm_kernel<<<kM, 256, 0, stream>>>(x, ln2g + l * kD, ln2b + l * kD, hb, kD);
    // mlp = gelu(h2 @ fc_w^T + fc_b) -> bf16
    gemm_bf16_kernel<2, 4><<<dim3(kMLP / 128, kM / 64, 1), 256, 0, stream>>>(
        hb, kD, 0, w_fc + (size_t)l * kMLP * kD, kD, 0,
        nullptr, mlpb, kMLP, 0, 0, 1,
        fcb + l * kMLP, nullptr, kM, kMLP, kD, /*gelu*/1, 0.f);
    // x += mlp @ fc2_w^T + fc2_b
    gemm_bf16_kernel<2, 4><<<dim3(kD / 128, kM / 64, 1), 256, 0, stream>>>(
        mlpb, kMLP, 0, w_fc2 + (size_t)l * kD * kMLP, kMLP, 0,
        x, nullptr, kD, 0, 0, 1,
        fc2b + l * kD, x, kM, kD, kMLP, 0, 0.f);
  }

  // final LN + LM head -> fp32 logits [M, V]
  layernorm_kernel<<<kM, 256, 0, stream>>>(x, lnfg, lnfb, hb, kD);
  gemm_bf16_kernel<2, 4><<<dim3(kV / 128, kM / 64, 1), 256, 0, stream>>>(
      hb, kD, 0, w_lm, kD, 0,
      logits, nullptr, kV, 0, 0, 1,
      nullptr, nullptr, kM, kV, kD, 0, 0.f);
}
